// MultiStrategyTokenGenerationHD_24764781429232
// MI455X (gfx1250) — compile-verified, matrix operand order checked by reading, not run on hardware
//
#include <hip/hip_runtime.h>
#include <hip/hip_fp16.h>

typedef float v2f __attribute__((ext_vector_type(2)));
typedef float v8f __attribute__((ext_vector_type(8)));
typedef _Float16 v16h __attribute__((ext_vector_type(16)));

#define BATCH 4
#define CF 512
#define NTOK 1024
#define IMG 512
#define KC 19
#define RROWS (KC * 256)   // 4864

#if __has_builtin(__builtin_amdgcn_wmma_f32_16x16x4_f32)
#define HAVE_WMMA_F32X4 1
#endif

// ---------------- token stats: mode/purity per 16x16 patch, conf-gated t ----
__global__ void k_tok_stats(const int* __restrict__ s_label,
                            const float* __restrict__ t_probs,
                            int* __restrict__ mode_s, int* __restrict__ pure_s,
                            int* __restrict__ mode_t, int* __restrict__ pure_t) {
  __shared__ int hs[20];
  __shared__ int ht[20];
  __shared__ float cred[256];
  int tid = threadIdx.x;
  int n = blockIdx.x, b = blockIdx.y;
  if (tid < 20) { hs[tid] = 0; ht[tid] = 0; }
  __syncthreads();
  int pi = tid >> 4, pj = tid & 15;
  int hp = n >> 5, wp = n & 31;
  int py = hp * 16 + pi, px = wp * 16 + pj;
  int sl = s_label[((size_t)b * IMG + py) * IMG + px];
  atomicAdd(&hs[sl + 1], 1);
  const float* tp = t_probs + (size_t)b * KC * IMG * IMG + (size_t)py * IMG + px;
  float mx = tp[0]; int am = 0;
  for (int k = 1; k < KC; ++k) {
    float v = tp[(size_t)k * IMG * IMG];
    if (v > mx) { mx = v; am = k; }
  }
  int tl = (mx < 0.9f) ? -1 : am;
  atomicAdd(&ht[tl + 1], 1);
  cred[tid] = mx;
  __syncthreads();
  for (int off = 128; off > 0; off >>= 1) {
    if (tid < off) cred[tid] += cred[tid + off];
    __syncthreads();
  }
  if (tid == 0) {
    int bi = 0, bc = hs[0];
    for (int i = 1; i < 20; ++i) if (hs[i] > bc) { bc = hs[i]; bi = i; }
    int ms = bi - 1;
    int nv = 256 - hs[0];
    int agree = (ms >= 0) ? hs[ms + 1] : 0;
    float pur = (float)agree / (float)(nv > 0 ? nv : 1);
    mode_s[b * NTOK + n] = ms;
    pure_s[b * NTOK + n] = (pur >= 0.9f) && (nv > 0);
    bi = 0; bc = ht[0];
    for (int i = 1; i < 20; ++i) if (ht[i] > bc) { bc = ht[i]; bi = i; }
    int mt = bi - 1;
    nv = 256 - ht[0];
    agree = (mt >= 0) ? ht[mt + 1] : 0;
    pur = (float)agree / (float)(nv > 0 ? nv : 1);
    int pt = (pur >= 0.9f) && (nv > 0);
    pt = pt && ((cred[0] * (1.0f / 256.0f)) >= 0.9f);
    mode_t[b * NTOK + n] = mt;
    pure_t[b * NTOK + n] = pt;
  }
}

// ---------------- bilinear 16x16 -> 32x32, tokenized [b,n,ch] ---------------
__global__ void k_resize(const float* __restrict__ sfm, const float* __restrict__ tfm,
                         float* __restrict__ stok, float* __restrict__ ttok) {
  int n = blockIdx.x, d = blockIdx.y, b = blockIdx.z;
  int ch = threadIdx.x;  // 512
  int hp = n >> 5, wp = n & 31;
  float fy = hp * 0.5f - 0.25f;
  float fx = wp * 0.5f - 0.25f;
  int y0 = (int)floorf(fy); float wy = fy - (float)y0;
  int x0 = (int)floorf(fx); float wx = fx - (float)x0;
  int y1 = y0 + 1, x1 = x0 + 1;
  y0 = y0 < 0 ? 0 : (y0 > 15 ? 15 : y0);
  y1 = y1 < 0 ? 0 : (y1 > 15 ? 15 : y1);
  x0 = x0 < 0 ? 0 : (x0 > 15 ? 15 : x0);
  x1 = x1 < 0 ? 0 : (x1 > 15 ? 15 : x1);
  const float* src = (d ? tfm : sfm) + ((size_t)b * CF + ch) * 256;
  float v00 = src[y0 * 16 + x0], v01 = src[y0 * 16 + x1];
  float v10 = src[y1 * 16 + x0], v11 = src[y1 * 16 + x1];
  float v = (1.f - wy) * ((1.f - wx) * v00 + wx * v01) +
            wy * ((1.f - wx) * v10 + wx * v11);
  float* dst = d ? ttok : stok;
  dst[((size_t)b * NTOK + n) * CF + ch] = v;
}

// ---------------- per-batch class counts + any-pure / any-impure flags ------
__global__ __launch_bounds__(1024) void k_counts(const int* mode_s, const int* pure_s,
                                                 const int* mode_t, const int* pure_t,
                                                 int* cnt, int* flags) {
  __shared__ int h0[19];
  __shared__ int h1[19];
  __shared__ int fl[4];
  int tid = threadIdx.x, b = blockIdx.x;
  if (tid < 19) { h0[tid] = 0; h1[tid] = 0; }
  if (tid < 4) fl[tid] = 0;
  __syncthreads();
  int ms = mode_s[b * NTOK + tid], ps = pure_s[b * NTOK + tid];
  int mt = mode_t[b * NTOK + tid], pt = pure_t[b * NTOK + tid];
  if (ps) { atomicAdd(&h0[ms], 1); atomicOr(&fl[0], 1); } else atomicOr(&fl[2], 1);
  if (pt) { atomicAdd(&h1[mt], 1); atomicOr(&fl[1], 1); } else atomicOr(&fl[3], 1);
  __syncthreads();
  if (tid < 19) {
    cnt[(b * 2 + 0) * 32 + tid] = h0[tid];
    cnt[(b * 2 + 1) * 32 + tid] = h1[tid];
  }
  if (tid < 4) flags[b * 4 + tid] = fl[tid];
}

// ---------------- WMMA: sums[c,ch] = sum_tok mask[c,tok]*feat[tok,ch] -------
__global__ void k_wmma_sums(const float* __restrict__ stok, const float* __restrict__ ttok,
                            const int* __restrict__ mode_s, const int* __restrict__ pure_s,
                            const int* __restrict__ mode_t, const int* __restrict__ pure_t,
                            float* __restrict__ sums) {
  __shared__ int cls[NTOK];
  int lane = threadIdx.x;      // 32 = one wave, EXEC all ones at the WMMA
  int mt = blockIdx.x >> 5;    // 0..1  (class tiles, 19 padded to 32)
  int nt = blockIdx.x & 31;    // 0..31 (channel tiles)
  int d = blockIdx.y, b = blockIdx.z;
  const float* feat = (d ? ttok : stok) + (size_t)b * NTOK * CF;
  const int* mode = (d ? mode_t : mode_s) + b * NTOK;
  const int* pure = (d ? pure_t : pure_s) + b * NTOK;
  for (int i = lane; i < NTOK; i += 32) cls[i] = pure[i] ? mode[i] : -1;
  __syncthreads();
  int half = lane >> 4, l15 = lane & 15;
  int myc = mt * 16 + l15;   // A-matrix row (class)
  int ch = nt * 16 + l15;    // B-matrix column (channel)
  v8f acc = {};
#ifdef HAVE_WMMA_F32X4
  // A 16x4 f32: VGPR0 = K(lane<16?0:2), VGPR1 = K(lane<16?1:3). B 4x16 mirrors.
  for (int k0 = 0; k0 < NTOK; k0 += 4) {
    int ka = k0 + half * 2;
    v2f A, Bv;
    A.x = (cls[ka] == myc) ? 1.f : 0.f;
    A.y = (cls[ka + 1] == myc) ? 1.f : 0.f;
    Bv.x = feat[(size_t)ka * CF + ch];
    Bv.y = feat[(size_t)(ka + 1) * CF + ch];
    acc = __builtin_amdgcn_wmma_f32_16x16x4_f32(false, A, false, Bv, (short)0,
                                                acc, false, false);
  }
#else
  // fallback: codegen-confirmed f16 WMMA (mask exact in f16; feat rounded)
  for (int k0 = 0; k0 < NTOK; k0 += 32) {
    v16h A, Bv;
#pragma unroll
    for (int j = 0; j < 16; ++j) {
      int kk = k0 + ((half == 0) ? (j < 8 ? j : j + 8) : (j < 8 ? j + 8 : j + 16));
      A[j] = (cls[kk] == myc) ? (_Float16)1.0f : (_Float16)0.0f;
      Bv[j] = (_Float16)feat[(size_t)kk * CF + ch];
    }
    acc = __builtin_amdgcn_wmma_f32_16x16x32_f16(false, A, false, Bv, (short)0,
                                                 acc, false, false);
  }
#endif
  // D layout: VGPR v -> class = mt*16 + v + (lane>=16 ? 8 : 0), ch column = l15
#pragma unroll
  for (int v = 0; v < 8; ++v) {
    int c = mt * 16 + v + half * 8;
    if (c < KC) sums[((size_t)(b * 2 + d) * 32 + c) * CF + ch] = acc[v];
  }
}

// ---------------- sequential momentum proto chain over batches --------------
__global__ void k_proto_seq(const float* __restrict__ sums, const int* __restrict__ cnt,
                            const int* __restrict__ flags, float* __restrict__ snap,
                            float* __restrict__ out_spro, float* __restrict__ out_tpro) {
  int ch = threadIdx.x;  // 512
  float ps[KC], pt[KC];
  for (int c = 0; c < KC; ++c) { ps[c] = 0.f; pt[c] = 0.f; }
  for (int b = 0; b < BATCH; ++b) {
    int aps = flags[b * 4 + 0], apt = flags[b * 4 + 1];
    int gs = aps, gst = aps && apt;
    for (int c = 0; c < KC; ++c) {
      int cs = cnt[(b * 2 + 0) * 32 + c];
      float mean = sums[((size_t)(b * 2 + 0) * 32 + c) * CF + ch] / (float)(cs > 0 ? cs : 1);
      if (gs && cs > 0) ps[c] = 0.99f * ps[c] + 0.01f * mean;
      snap[((size_t)(b * 2 + 0) * 32 + c) * CF + ch] = ps[c];
      int ct = cnt[(b * 2 + 1) * 32 + c];
      mean = sums[((size_t)(b * 2 + 1) * 32 + c) * CF + ch] / (float)(ct > 0 ? ct : 1);
      if (gst && ct > 0) pt[c] = 0.99f * pt[c] + 0.01f * mean;
      snap[((size_t)(b * 2 + 1) * 32 + c) * CF + ch] = pt[c];
    }
  }
  for (int c = 0; c < KC; ++c) {
    out_spro[c * CF + ch] = ps[c];
    out_tpro[c * CF + ch] = pt[c];
  }
}

// ---------------- distances ||feat - proto[max(mode,0)]|| (wave per token) --
__global__ void k_dist(const float* __restrict__ stok, const float* __restrict__ ttok,
                       const int* __restrict__ mode_s, const int* __restrict__ mode_t,
                       const float* __restrict__ snap, float* __restrict__ dist) {
  int d = blockIdx.y, b = blockIdx.z;
  int tid = threadIdx.x;
  int lane = tid & 31, w = tid >> 5;
  int n = blockIdx.x * 8 + w;
  const float* feat = (d ? ttok : stok) + ((size_t)b * NTOK + n) * CF;
  int m = (d ? mode_t : mode_s)[b * NTOK + n];
  if (m < 0) m = 0;
  const float* pr = snap + ((size_t)(b * 2 + d) * 32 + m) * CF;
  float s = 0.f;
  for (int i = 0; i < 16; ++i) {
    int ch = lane + i * 32;
    float df = feat[ch] - pr[ch];
    s += df * df;
  }
  for (int off = 16; off > 0; off >>= 1) s += __shfl_down(s, off, 32);
  if (lane == 0) dist[(size_t)(b * 2 + d) * NTOK + n] = sqrtf(s);
}

// ---------------- per-(class,dir): sort sources by dist, cyclic assignment --
__global__ __launch_bounds__(1024) void k_match(const int* mode_s, const int* pure_s,
                                                const int* mode_t, const int* pure_t,
                                                const float* __restrict__ dist,
                                                int* __restrict__ srcmap,
                                                int* __restrict__ okmap) {
  __shared__ float key[NTOK];
  __shared__ int sid[NTOK];
  __shared__ int sc[NTOK];
  __shared__ int nsrc;
  int c = blockIdx.x, d = blockIdx.y, b = blockIdx.z, tid = threadIdx.x;
  int srcDom = d ^ 1;  // d=0: tgt=s src=t ; d=1: tgt=t src=s
  const int* sm_ = srcDom ? mode_t : mode_s;
  const int* sp_ = srcDom ? pure_t : pure_s;
  const int* tm_ = d ? mode_t : mode_s;
  const int* tp_ = d ? pure_t : pure_s;
  int msrc = (sp_[b * NTOK + tid] && sm_[b * NTOK + tid] == c) ? 1 : 0;
  key[tid] = msrc ? dist[(size_t)(b * 2 + srcDom) * NTOK + tid] : __builtin_inff();
  sid[tid] = tid;
  if (tid == 0) nsrc = 0;
  __syncthreads();
  atomicAdd(&nsrc, msrc);
  // bitonic sort ascending by (key, id) -> stable argsort semantics
  for (int kk = 2; kk <= NTOK; kk <<= 1) {
    for (int j = kk >> 1; j > 0; j >>= 1) {
      __syncthreads();
      int ixj = tid ^ j;
      if (ixj > tid) {
        float ka = key[tid], kb = key[ixj];
        int ia = sid[tid], ib = sid[ixj];
        bool asc = ((tid & kk) == 0);
        bool sw = asc ? (ka > kb || (ka == kb && ia > ib))
                      : (ka < kb || (ka == kb && ia < ib));
        if (sw) { key[tid] = kb; key[ixj] = ka; sid[tid] = ib; sid[ixj] = ia; }
      }
    }
  }
  __syncthreads();
  int mtgt = (tp_[b * NTOK + tid] && tm_[b * NTOK + tid] == c) ? 1 : 0;
  sc[tid] = mtgt;
  __syncthreads();
  for (int off = 1; off < NTOK; off <<= 1) {
    int v = (tid >= off) ? sc[tid - off] : 0;
    __syncthreads();
    sc[tid] += v;
    __syncthreads();
  }
  if (mtgt) {
    int ns = nsrc;
    int rank = sc[tid] - 1;
    int idx = rank % (ns > 0 ? ns : 1);
    srcmap[(size_t)(b * 2 + d) * NTOK + tid] = sid[idx];
    okmap[(size_t)(b * 2 + d) * NTOK + tid] = (ns > 0);
  }
}

// ---------------- masked mean/std over impure tokens ------------------------
__global__ void k_impstats(const float* __restrict__ stok, const float* __restrict__ ttok,
                           const int* __restrict__ pure_s, const int* __restrict__ pure_t,
                           float* __restrict__ istat) {
  int chunk = blockIdx.x, d = blockIdx.y, b = blockIdx.z;
  int ch = chunk * 256 + threadIdx.x;
  const float* feat = (d ? ttok : stok) + (size_t)b * NTOK * CF;
  const int* pure = (d ? pure_t : pure_s) + b * NTOK;
  float s = 0.f, s2 = 0.f;
  int cn = 0;
  for (int t = 0; t < NTOK; ++t) {
    if (!pure[t]) {
      float v = feat[(size_t)t * CF + ch];
      s += v; s2 += v * v; cn++;
    }
  }
  float nf = (float)(cn > 0 ? cn : 1);
  float mean = s / nf;
  float var = (cn > 1) ? (s2 - s * s / nf) / (float)(cn - 1) : 0.f;
  var = var > 0.f ? var : 0.f;
  float sd = (cn > 1) ? sqrtf(var) : 0.f;
  istat[((size_t)(b * 2 + d) * 2 + 0) * CF + ch] = mean;
  istat[((size_t)(b * 2 + d) * 2 + 1) * CF + ch] = sd;
}

// ---------------- feature composition ---------------------------------------
__global__ void k_feat(const float* __restrict__ stok, const float* __restrict__ ttok,
                       const int* __restrict__ pure_s, const int* __restrict__ pure_t,
                       const int* __restrict__ srcmap, const int* __restrict__ okmap,
                       const int* __restrict__ flags, const float* __restrict__ istat,
                       const float* __restrict__ alphas,
                       float* __restrict__ out_sfc, float* __restrict__ out_tfc) {
  int n = blockIdx.x, b = blockIdx.y, tid = threadIdx.x;  // 256
  float alpha = alphas[b];
  int gst = flags[b * 4 + 0] && flags[b * 4 + 1];
  int both = flags[b * 4 + 2] && flags[b * 4 + 3];
  int ps = pure_s[b * NTOK + n], pt = pure_t[b * NTOK + n];
  int oks = okmap[(size_t)(b * 2 + 0) * NTOK + n];
  int srcs = srcmap[(size_t)(b * 2 + 0) * NTOK + n];
  int okt = okmap[(size_t)(b * 2 + 1) * NTOK + n];
  int srct = srcmap[(size_t)(b * 2 + 1) * NTOK + n];
  bool apS = ps && gst && oks;
  bool apT = pt && gst && okt;
  int sS = apS ? srcs : 0;
  int sT = apT ? srct : 0;
  const float* ist = istat + (size_t)b * 4 * CF;  // [smean,sstd,tmean,tstd][CF]
  for (int j = 0; j < 2; ++j) {
    int ch = tid + j * 256;
    float sf = stok[((size_t)b * NTOK + n) * CF + ch];
    float tf = ttok[((size_t)b * NTOK + n) * CF + ch];
    float sm = ist[0 * CF + ch], ss = ist[1 * CF + ch];
    float tm = ist[2 * CF + ch], ts = ist[3 * CF + ch];
    float so, to;
    if (apS) so = alpha * sf + (1.f - alpha) * ttok[((size_t)b * NTOK + sS) * CF + ch];
    else if (!ps && both) {
      float styl = (sf - sm) / (ss + 1e-5f) * ts + tm;
      so = alpha * sf + (1.f - alpha) * styl;
    } else so = sf;
    if (apT) to = alpha * tf + (1.f - alpha) * stok[((size_t)b * NTOK + sT) * CF + ch];
    else if (!pt && both) {
      float styl = (tf - tm) / (ts + 1e-5f) * ss + sm;
      to = alpha * tf + (1.f - alpha) * styl;
    } else to = tf;
    out_sfc[((size_t)b * NTOK + n) * CF + ch] = so;
    out_tfc[((size_t)b * NTOK + n) * CF + ch] = to;
  }
}

// ---------------- label composition (bandwidth-dominant) --------------------
__global__ __launch_bounds__(1024) void k_label(const int* __restrict__ s_label,
                                                const float* __restrict__ t_probs,
                                                const int* __restrict__ pure_s,
                                                const int* __restrict__ pure_t,
                                                const int* __restrict__ srcmap,
                                                const int* __restrict__ okmap,
                                                const int* __restrict__ flags,
                                                float* __restrict__ out_slc,
                                                float* __restrict__ out_tlc) {
  int r = blockIdx.x, b = blockIdx.y, n = threadIdx.x;
  int k = r >> 8;
  int p = r & 255;
  int pi = p >> 4, pj = p & 15;
  int gst = flags[b * 4 + 0] && flags[b * 4 + 1];
  int hp = n >> 5, wp = n & 31;
  int py = hp * 16 + pi, px = wp * 16 + pj;
  bool apS = pure_s[b * NTOK + n] && gst && okmap[(size_t)(b * 2 + 0) * NTOK + n];
  bool apT = pure_t[b * NTOK + n] && gst && okmap[(size_t)(b * 2 + 1) * NTOK + n];
  const float* tpk = t_probs + ((size_t)b * KC + k) * IMG * IMG;
  const int* slb = s_label + (size_t)b * IMG * IMG;
  float sval, tval;
  if (apS) {
    int n2 = srcmap[(size_t)(b * 2 + 0) * NTOK + n];
    int qy = (n2 >> 5) * 16 + pi, qx = (n2 & 31) * 16 + pj;
    sval = tpk[(size_t)qy * IMG + qx];  // PSEUDO_W == 1.0
  } else {
    sval = (slb[(size_t)py * IMG + px] == k) ? 1.f : 0.f;
  }
  if (apT) {
    int n2 = srcmap[(size_t)(b * 2 + 1) * NTOK + n];
    int qy = (n2 >> 5) * 16 + pi, qx = (n2 & 31) * 16 + pj;
    tval = (slb[(size_t)qy * IMG + qx] == k) ? 1.f : 0.f;
  } else {
    tval = tpk[(size_t)py * IMG + px];
  }
  out_slc[((size_t)b * RROWS + r) * NTOK + n] = sval;
  out_tlc[((size_t)b * RROWS + r) * NTOK + n] = tval;
}

extern "C" void kernel_launch(void* const* d_in, const int* in_sizes, int n_in,
                              void* d_out, int out_size, void* d_ws, size_t ws_size,
                              hipStream_t stream) {
  const float* sfm = (const float*)d_in[0];
  const float* tfm = (const float*)d_in[1];
  const int* slab = (const int*)d_in[2];
  const float* tpr = (const float*)d_in[3];
  const float* alph = (const float*)d_in[4];
  float* out = (float*)d_out;

  char* ws = (char*)d_ws;
  float* stok   = (float*)(ws + 0);         // 8 MB
  float* ttok   = (float*)(ws + 8388608);   // 8 MB
  int*   mode_s = (int*)(ws + 16777216);
  int*   pure_s = (int*)(ws + 16793600);
  int*   mode_t = (int*)(ws + 16809984);
  int*   pure_t = (int*)(ws + 16826368);
  float* sums   = (float*)(ws + 16842752);  // [B][2][32][CF]
  int*   cnt    = (int*)(ws + 17367040);    // [B][2][32]
  int*   flags  = (int*)(ws + 17368064);    // [B][4]
  float* snap   = (float*)(ws + 17368128);  // [B][2][32][CF]
  float* dist   = (float*)(ws + 17892416);  // [B][2][NTOK]
  int*   srcmap = (int*)(ws + 17925184);    // [B][2][NTOK]
  int*   okmap  = (int*)(ws + 17957952);    // [B][2][NTOK]
  float* istat  = (float*)(ws + 17990720);  // [B][2][2][CF]

  float* out_sfc = out;
  float* out_tfc = out + 2097152;
  float* out_slc = out + 4194304;
  float* out_tlc = out + 24117248;
  float* out_spro = out + 44040192;
  float* out_tpro = out + 44049920;

  k_tok_stats<<<dim3(NTOK, BATCH), 256, 0, stream>>>(slab, tpr, mode_s, pure_s, mode_t, pure_t);
  k_resize<<<dim3(NTOK, 2, BATCH), 512, 0, stream>>>(sfm, tfm, stok, ttok);
  k_counts<<<BATCH, 1024, 0, stream>>>(mode_s, pure_s, mode_t, pure_t, cnt, flags);
  k_wmma_sums<<<dim3(64, 2, BATCH), 32, 0, stream>>>(stok, ttok, mode_s, pure_s, mode_t, pure_t, sums);
  k_proto_seq<<<1, 512, 0, stream>>>(sums, cnt, flags, snap, out_spro, out_tpro);
  k_dist<<<dim3(NTOK / 8, 2, BATCH), 256, 0, stream>>>(stok, ttok, mode_s, mode_t, snap, dist);
  k_match<<<dim3(KC, 2, BATCH), 1024, 0, stream>>>(mode_s, pure_s, mode_t, pure_t, dist, srcmap, okmap);
  k_impstats<<<dim3(2, 2, BATCH), 256, 0, stream>>>(stok, ttok, pure_s, pure_t, istat);
  k_feat<<<dim3(NTOK, BATCH), 256, 0, stream>>>(stok, ttok, pure_s, pure_t, srcmap, okmap, flags,
                                                istat, alph, out_sfc, out_tfc);
  k_label<<<dim3(RROWS, BATCH), 1024, 0, stream>>>(slab, tpr, pure_s, pure_t, srcmap, okmap, flags,
                                                   out_slc, out_tlc);
}
